// WaveNet_58926951301188
// MI455X (gfx1250) — matrix operand order repaired and checked by reading, not compiled, not run on hardware
//
#include <hip/hip_runtime.h>
#include <hip/hip_bf16.h>
#include <math.h>

// ---------------------------------------------------------------------------
// WaveNet on gfx1250 (MI455X): all channel mixing via V_WMMA_F32_16X16X32_F16.
// C=16 -> every conv is a 16x16 tile with K=32 contraction (16 cin x 2 taps).
// Weights are pre-packed once into WMMA A-operand order (f16) so each wave
// loads operands as single 32B vectors. Input tiles are staged to LDS with
// GLOBAL_LOAD_ASYNC_TO_LDS_B128 when the toolchain exposes the builtin.
// Working set (~24 MB ping-pong + skip) is L2-resident (192 MB L2).
// ---------------------------------------------------------------------------

typedef __attribute__((ext_vector_type(16))) _Float16 v16h;
typedef __attribute__((ext_vector_type(8)))  float    v8f;
typedef int v4i_ __attribute__((vector_size(16)));

#define GLOBAL_AS __attribute__((address_space(1)))
#define LDS_AS    __attribute__((address_space(3)))

#define BB    8
#define NN    16384
#define CC    16
#define LL    30
#define AGGC  64
#define OUTC  256
#define HALO  512            // max dilation 2^9
#define TILE_T 128           // per block: 8 waves x 16 timesteps
#define XW   (HALO + TILE_T) // staged LDS row width (640)

#if defined(__HIP_DEVICE_COMPILE__) && \
    __has_builtin(__builtin_amdgcn_global_load_async_to_lds_b128) && \
    __has_builtin(__builtin_amdgcn_s_wait_asynccnt)
#define HAVE_ASYNC_LDS 1
#else
#define HAVE_ASYNC_LDS 0
#endif

// Pack geometry (v16h units): per gated layer 3 matrices x 32 lanes,
// then 4 Wa tiles, then 16x2 Wo chunks.
#define PK_GATED_V ((LL - 1) * 3 * 32)
#define PK_WA_V    (4 * 32)
#define PK_WO_V    (32 * 32)
#define PK_TOTAL_H ((PK_GATED_V + PK_WA_V + PK_WO_V) * 16)

__device__ __forceinline__ float fast_sigmoid(float v) {
    return __builtin_amdgcn_rcpf(1.0f + __expf(-v));
}
__device__ __forceinline__ float fast_tanh(float v) {
    // tanh(v) = 2*sigmoid(2v) - 1  (branch-free: v_exp + v_rcp)
    return 2.0f * __builtin_amdgcn_rcpf(1.0f + __expf(-2.0f * v)) - 1.0f;
}

// ---------------------------------------------------------------------------
// Pre-pack all weights as f16 WMMA A operands in contiguous k order (B matches):
// lane m = l&15, half = l>>4, element j -> K = j + 16*half.
// ---------------------------------------------------------------------------
__global__ __launch_bounds__(256) void k_pack(const float* __restrict__ Wf,
                                              const float* __restrict__ Wg,
                                              const float* __restrict__ Wr,
                                              const float* __restrict__ Wa,
                                              const float* __restrict__ Wo,
                                              _Float16* __restrict__ pk) {
    for (int idx = blockIdx.x * 256 + threadIdx.x; idx < PK_TOTAL_H;
         idx += gridDim.x * 256) {
        int elem = idx & 15;
        int lane = (idx >> 4) & 31;
        int grp  = idx >> 9;
        int half = lane >> 4, m = lane & 15;
        int q = elem + 16 * half;
        float v;
        if (grp < (LL - 1) * 3) {                       // gated-layer weights
            int j = grp / 3, mat = grp - 3 * j;
            size_t wb = (size_t)j * CC * CC;
            if (mat == 0)      v = Wf[(wb + m * CC + (q & 15)) * 2 + (q >> 4)];
            else if (mat == 1) v = Wg[(wb + m * CC + (q & 15)) * 2 + (q >> 4)];
            else               v = (q < CC) ? Wr[wb + m * CC + q] : 0.0f;
        } else if (grp < (LL - 1) * 3 + 4) {            // Wa row tiles (K-pad)
            int mt = grp - (LL - 1) * 3;
            v = (q < CC) ? Wa[(mt * 16 + m) * CC + q] : 0.0f;
        } else {                                        // Wo (ot, kc) chunks
            int g2 = grp - ((LL - 1) * 3 + 4);
            int ot = g2 >> 1, kc = g2 & 1;
            v = Wo[(ot * 16 + m) * AGGC + kc * 32 + q];
        }
        pk[idx] = (_Float16)v;
    }
}

// ---------------------------------------------------------------------------
// Layer 0: plain causal conv 1->16 (K=2, d=1), skip init, residual broadcast.
// ---------------------------------------------------------------------------
__global__ __launch_bounds__(256) void k_layer0(const float* __restrict__ x,
                                                const float* __restrict__ W0,
                                                const float* __restrict__ b0,
                                                float* __restrict__ inp,
                                                float* __restrict__ skp) {
    int gid = blockIdx.x * 256 + threadIdx.x;           // over B*N
    int b = gid / NN, t = gid % NN;
    const float s = 1.0f / 32768.0f;
    float xt = x[(size_t)b * NN + t] * s;
    float xp = (t > 0) ? x[(size_t)b * NN + t - 1] * s : 0.0f;
#pragma unroll
    for (int c = 0; c < CC; ++c) {
        float h = W0[c * 2 + 0] * xp + W0[c * 2 + 1] * xt + b0[c];
        size_t idx = ((size_t)b * CC + c) * NN + t;
        skp[idx] = h;
        inp[idx] = xt + h;
    }
}

// ---------------------------------------------------------------------------
// Gated layer:  f = tanh(conv(inp,Wf,d)+bf); g = sigmoid(conv(inp,Wg,d)+bg);
//               out = Wr @ (f*g) + br;  skip += out;  inp_next = inp + out.
// One wave owns a 16x16 (channels x time) output tile: 3 WMMAs per wave.
// ---------------------------------------------------------------------------
__global__ __launch_bounds__(256) void k_gated(const float* __restrict__ src,
                                               float* __restrict__ dst,
                                               float* __restrict__ skp,
                                               const _Float16* __restrict__ pkL,
                                               const float* __restrict__ bf,
                                               const float* __restrict__ bg,
                                               const float* __restrict__ br,
                                               int d) {
    __shared__ float    xs[CC * XW];     // staged f32 input tile + halo (40 KB)
    __shared__ _Float16 zs[8 * 256];     // per-wave 16x16 gated tile (4 KB)

    const int tid = threadIdx.x;
    const int b  = blockIdx.y;
    const int t0 = blockIdx.x * TILE_T;
    const float* srcB = src + (size_t)b * CC * NN;

    // ---- Stage input tile (+halo, causal zero pad) into LDS ----
#if HAVE_ASYNC_LDS
    for (int e4 = tid; e4 < CC * XW / 4; e4 += 256) {
        int c  = e4 / (XW / 4);
        int j4 = (e4 - c * (XW / 4)) * 4;
        int tg = t0 - HALO + j4;                         // chunk fully in/out
        float* l = &xs[c * XW + j4];
        if (tg >= 0) {
            __builtin_amdgcn_global_load_async_to_lds_b128(
                (GLOBAL_AS v4i_*)(srcB + (size_t)c * NN + tg),
                (LDS_AS v4i_*)l, 0, 0);
        } else {
            l[0] = 0.0f; l[1] = 0.0f; l[2] = 0.0f; l[3] = 0.0f;
        }
    }
    __builtin_amdgcn_s_wait_asynccnt(0);
#else
    for (int e = tid; e < CC * XW; e += 256) {
        int c = e / XW, j = e - c * XW;
        int tg = t0 - HALO + j;
        xs[c * XW + j] = (tg >= 0) ? srcB[(size_t)c * NN + tg] : 0.0f;
    }
#endif
    __syncthreads();

    const int lane = tid & 31;
    const int wave = tid >> 5;
    const int half = lane >> 4;
    const int n    = lane & 15;          // WMMA B/C column (time within tile)
    const int tw   = wave * 16;

    // A operands: one 32B vector load each from the pre-packed buffer.
    const v16h* pkv = (const v16h*)pkL;
    v16h Af = pkv[0 * 32 + lane];
    v16h Ag = pkv[1 * 32 + lane];
    v16h Ar = pkv[2 * 32 + lane];

    // B operand (32x16 f16): K = j + 16*half -> tap==half, cin==j.
    v16h Bx;
    const int jj = HALO + tw + n - (half ? 0 : d);
#pragma unroll
    for (int j = 0; j < 16; ++j) Bx[j] = (_Float16)xs[j * XW + jj];

    v8f zero = {};
    v8f accF = __builtin_amdgcn_wmma_f32_16x16x32_f16(false, Af, false, Bx,
                                                      (short)0, zero, false, false);
    v8f accG = __builtin_amdgcn_wmma_f32_16x16x32_f16(false, Ag, false, Bx,
                                                      (short)0, zero, false, false);

    // Gated activation (branch-free), re-pack 16x16 tile via LDS transpose.
#pragma unroll
    for (int r = 0; r < 8; ++r) {
        int mr = r + 8 * half;
        float f = fast_tanh(accF[r] + bf[mr]);
        float g = fast_sigmoid(accG[r] + bg[mr]);
        zs[wave * 256 + mr * 16 + n] = (_Float16)(f * g);
    }
    __builtin_amdgcn_wave_barrier();
    asm volatile("s_wait_dscnt 0" ::: "memory");         // wave-local DS RAW
    __builtin_amdgcn_wave_barrier();

    v16h zB;                                             // K(=cin) 0..15, pad up
#pragma unroll
    for (int j = 0; j < 16; ++j)
        zB[j] = half ? (_Float16)0.0f : zs[wave * 256 + j * 16 + n];

    v8f accO = __builtin_amdgcn_wmma_f32_16x16x32_f16(false, Ar, false, zB,
                                                      (short)0, zero, false, false);

    // Residual + skip updates (owned (channel,t) cells only -> race free).
    const int t = t0 + tw + n;
    float* dstB = dst + (size_t)b * CC * NN;
    float* skpB = skp + (size_t)b * CC * NN;
#pragma unroll
    for (int r = 0; r < 8; ++r) {
        int mr = r + 8 * half;
        size_t idx = (size_t)mr * NN + t;
        float o = accO[r] + br[mr];
        dstB[idx] = srcB[idx] + o;
        skpB[idx] = skpB[idx] + o;
    }
}

// ---------------------------------------------------------------------------
// Head: a = relu(Wa @ relu(skip) + ba);  o = Wo @ a + bo;  log_softmax; mask.
// Two passes over the 16 Wo row tiles with branch-free online logsumexp;
// half-lanes combined with __shfl_xor(.,16) (wave32).
// ---------------------------------------------------------------------------
__global__ __launch_bounds__(256) void k_head(const float* __restrict__ skp,
                                              const int* __restrict__ lengths,
                                              const _Float16* __restrict__ pkH,
                                              const float* __restrict__ ba,
                                              const float* __restrict__ bo,
                                              float* __restrict__ outp) {
    __shared__ _Float16 aS[8 * AGGC * 16];               // 8 waves x 64 x 16

    const int tid  = threadIdx.x;
    const int lane = tid & 31;
    const int wave = tid >> 5;
    const int half = lane >> 4;
    const int n    = lane & 15;
    const int b    = blockIdx.y;
    const int t    = blockIdx.x * 128 + wave * 16 + n;
    const float* skpB = skp + (size_t)b * CC * NN;
    const v16h* pkv = (const v16h*)pkH;                  // [4 Wa | 32 Wo] v16h

    // B operand: relu(skip) as 16(K, padded to 32) x 16(time).
    v16h sB;
#pragma unroll
    for (int j = 0; j < 16; ++j) {
        float v = half ? 0.0f : fmaxf(skpB[(size_t)j * NN + t], 0.0f);
        sB[j] = (_Float16)v;
    }

    v8f zero = {};
    // a = relu(Wa @ relu(skip) + ba): 4 row tiles of 16.
#pragma unroll
    for (int mt = 0; mt < 4; ++mt) {
        v16h Aw = pkv[mt * 32 + lane];
        v8f acc = __builtin_amdgcn_wmma_f32_16x16x32_f16(false, Aw, false, sB,
                                                         (short)0, zero, false, false);
#pragma unroll
        for (int r = 0; r < 8; ++r) {
            int ach = mt * 16 + r + 8 * half;
            aS[wave * AGGC * 16 + ach * 16 + n] =
                (_Float16)fmaxf(acc[r] + ba[ach], 0.0f);
        }
    }
    __builtin_amdgcn_wave_barrier();
    asm volatile("s_wait_dscnt 0" ::: "memory");
    __builtin_amdgcn_wave_barrier();

    // B operands for Wo: K=64 in two 32-chunks.
    v16h aB0, aB1;
#pragma unroll
    for (int j = 0; j < 16; ++j) {
        int k0 = j + 16 * half;
        aB0[j] = aS[wave * AGGC * 16 + k0 * 16 + n];
        aB1[j] = aS[wave * AGGC * 16 + (32 + k0) * 16 + n];
    }

    // Pass 1: branch-free online logsumexp over this lane's 128 rows.
    float mx = -__builtin_inff(), sm = 0.0f;
#pragma unroll 4
    for (int ot = 0; ot < 16; ++ot) {
        v16h A0 = pkv[128 + (ot * 2 + 0) * 32 + lane];
        v16h A1 = pkv[128 + (ot * 2 + 1) * 32 + lane];
        v8f acc = __builtin_amdgcn_wmma_f32_16x16x32_f16(false, A0, false, aB0,
                                                         (short)0, zero, false, false);
        acc = __builtin_amdgcn_wmma_f32_16x16x32_f16(false, A1, false, aB1,
                                                     (short)0, acc, false, false);
#pragma unroll
        for (int r = 0; r < 8; ++r) {
            float v = acc[r] + bo[ot * 16 + r + 8 * half];
            float nm = fmaxf(mx, v);
            sm = sm * __expf(mx - nm) + __expf(v - nm);
            mx = nm;
        }
    }
    // Combine the two half-lanes holding the same time column (wave32).
    float mx2 = __shfl_xor(mx, 16);
    float sm2 = __shfl_xor(sm, 16);
    float M = fmaxf(mx, mx2);
    float S = sm * __expf(mx - M) + sm2 * __expf(mx2 - M);
    float logZ = M + __logf(S);

    const int  len   = lengths[b];
    const bool valid = (t < len);
    float* outB = outp + (size_t)b * OUTC * NN;

    // Pass 2: recompute tiles, write masked log-probs.
#pragma unroll 4
    for (int ot = 0; ot < 16; ++ot) {
        v16h A0 = pkv[128 + (ot * 2 + 0) * 32 + lane];
        v16h A1 = pkv[128 + (ot * 2 + 1) * 32 + lane];
        v8f acc = __builtin_amdgcn_wmma_f32_16x16x32_f16(false, A0, false, aB0,
                                                         (short)0, zero, false, false);
        acc = __builtin_amdgcn_wmma_f32_16x16x32_f16(false, A1, false, aB1,
                                                     (short)0, acc, false, false);
#pragma unroll
        for (int r = 0; r < 8; ++r) {
            int row = ot * 16 + r + 8 * half;
            float v = acc[r] + bo[row];
            outB[(size_t)row * NN + t] = valid ? (v - logZ) : 0.0f;
        }
    }
}

// ---------------------------------------------------------------------------
// Host launcher: pack + 1 + 29 + 1 stream-ordered launches. `inp` ping-pongs
// in workspace to avoid the in-place halo race across time tiles.
// ---------------------------------------------------------------------------
extern "C" void kernel_launch(void* const* d_in, const int* in_sizes, int n_in,
                              void* d_out, int out_size, void* d_ws, size_t ws_size,
                              hipStream_t stream) {
    const float* x       = (const float*)d_in[0];
    const int*   lengths = (const int*)d_in[1];
    const float* W0 = (const float*)d_in[2];
    const float* b0 = (const float*)d_in[3];
    const float* Wf = (const float*)d_in[4];
    const float* bf = (const float*)d_in[5];
    const float* Wg = (const float*)d_in[6];
    const float* bg = (const float*)d_in[7];
    const float* Wr = (const float*)d_in[8];
    const float* br = (const float*)d_in[9];
    const float* Wa = (const float*)d_in[10];
    const float* ba = (const float*)d_in[11];
    const float* Wo = (const float*)d_in[12];
    const float* bo = (const float*)d_in[13];
    float* outp = (float*)d_out;

    float* ws = (float*)d_ws;
    const size_t BCN = (size_t)BB * CC * NN;    // 2M floats
    float* inp0 = ws;                           // ping
    float* inp1 = ws + BCN;                     // pong
    float* skp  = ws + 2 * BCN;                 // skip accumulator
    _Float16* pack = (_Float16*)(ws + 3 * BCN); // packed WMMA A operands
    (void)ws_size; (void)in_sizes; (void)n_in; (void)out_size;

    k_pack<<<dim3(64), dim3(256), 0, stream>>>(Wf, Wg, Wr, Wa, Wo, pack);
    k_layer0<<<dim3((BB * NN) / 256), dim3(256), 0, stream>>>(x, W0, b0, inp0, skp);

    for (int i = 1; i < LL; ++i) {
        int j = i - 1;
        int d = 1 << (i % 10);
        const float* src = (j & 1) ? inp1 : inp0;
        float*       dst = (j & 1) ? inp0 : inp1;
        k_gated<<<dim3(NN / TILE_T, BB), dim3(256), 0, stream>>>(
            src, dst, skp,
            pack + (size_t)j * 3 * 32 * 16,
            bf + j * CC, bg + j * CC, br + j * CC, d);
    }

    k_head<<<dim3(NN / 128, BB), dim3(256), 0, stream>>>(
        skp, lengths, pack + (size_t)PK_GATED_V * 16, ba, bo, outp);
}
